// MultiHeadAttentionWithKV_45019847197307
// MI455X (gfx1250) — matrix operand order repaired and checked by reading, not compiled, not run on hardware
//
#include <hip/hip_runtime.h>

#define B_    8
#define SQ_   16
#define HID_  1024
#define NH_   16
#define DH_   64
#define PAST_ 4096
#define SKV_  4112
#define MROWS_ 128
#define NSEG_ 8

typedef __bf16 bf16;
typedef __attribute__((ext_vector_type(16))) __bf16 v16bf;
typedef __attribute__((ext_vector_type(8)))  __bf16 v8bf;
typedef __attribute__((ext_vector_type(8)))  float  v8f;

// ---------------------------------------------------------------- conversions
__global__ void cvt_hidden(const float* __restrict__ h, bf16* __restrict__ o, int n) {
    int i = blockIdx.x * blockDim.x + threadIdx.x;
    if (i < n) o[i] = (bf16)h[i];
}

// transpose+convert W (k-major) -> Wt[n][k] bf16, one matrix per blockIdx.y
__global__ void cvt_wT(const float* __restrict__ w0, const float* __restrict__ w1,
                       const float* __restrict__ w2, const float* __restrict__ w3,
                       bf16* __restrict__ o0, bf16* __restrict__ o1,
                       bf16* __restrict__ o2, bf16* __restrict__ o3) {
    const float* w; bf16* o;
    switch (blockIdx.y) {
        case 0:  w = w0; o = o0; break;
        case 1:  w = w1; o = o1; break;
        case 2:  w = w2; o = o2; break;
        default: w = w3; o = o3; break;
    }
    int i = blockIdx.x * blockDim.x + threadIdx.x;   // 0..HID*HID-1
    int n = i & (HID_ - 1);
    int k = i >> 10;
    o[(size_t)n * HID_ + k] = (bf16)w[(size_t)k * HID_ + n];
}

// -------------------------------------------------------- bf16 WMMA GEMM
// C[128 x 1024] = A[128 x 1024] * Bt^T + bias ; one wave per 16x16 tile.
__global__ void __launch_bounds__(32)
gemm_bf16(const bf16* __restrict__ A, const bf16* __restrict__ Bt,
          const float* __restrict__ bias, float* __restrict__ C) {
    int tile = blockIdx.x;                 // 8 row-tiles * 64 col-tiles
    int m0 = (tile >> 6) << 4;
    int n0 = (tile & 63) << 4;
    int lane = threadIdx.x;
    int half = lane >> 4;                  // 0/1
    int l16  = lane & 15;
    const bf16* arow = A  + (size_t)(m0 + l16) * HID_;
    const bf16* bcol = Bt + (size_t)(n0 + l16) * HID_;
    v8f c = {};
    for (int kk = 0; kk < HID_; kk += 32) {
        union { v16bf v; v8bf h[2]; } a, b;
        a.h[0] = *(const v8bf*)(arow + kk + half * 8);        // K = h*8 .. +7
        a.h[1] = *(const v8bf*)(arow + kk + 16 + half * 8);   // K = 16+h*8 ..
        b.h[0] = *(const v8bf*)(bcol + kk + half * 8);        // K = h*8 .. +7
        b.h[1] = *(const v8bf*)(bcol + kk + 16 + half * 8);   // K = 16+h*8 ..
        c = __builtin_amdgcn_wmma_f32_16x16x32_bf16(false, a.v, false, b.v,
                                                    (short)0, c, false, false);
    }
    float bv = bias[n0 + l16];
    float* crow = C + (size_t)(m0 + half * 8) * HID_ + n0 + l16;
#pragma unroll
    for (int r = 0; r < 8; ++r) crow[(size_t)r * HID_] = c[r] + bv;
}

// ---------------------------------------------------- RoPE + KV scatter
__global__ void rope_scatter(const float* __restrict__ Qraw, const float* __restrict__ Kraw,
                             const float* __restrict__ Vraw, const int* __restrict__ pos,
                             bf16* __restrict__ qbf, float* __restrict__ newK,
                             float* __restrict__ newV) {
    int i = blockIdx.x * blockDim.x + threadIdx.x;     // B*SQ*512
    if (i >= B_ * SQ_ * 512) return;
    int j   = i & 511;
    int row = i >> 9;                                   // b*16+s
    int b = row >> 4, s = row & 15;
    float p   = (float)pos[row];
    float inv = __expf(-(float)j * (9.210340371976184f / 512.0f)); // 10000^(-j/512)
    float sn, cs;
    __sincosf(p * inv, &sn, &cs);
    const float* qr = Qraw + (size_t)row * HID_;
    const float* kr = Kraw + (size_t)row * HID_;
    const float* vr = Vraw + (size_t)row * HID_;
    float q1 = qr[j], q2 = qr[j + 512];
    float k1 = kr[j], k2 = kr[j + 512];
    float qo1 = q1 * cs - q2 * sn, qo2 = q2 * cs + q1 * sn;
    float ko1 = k1 * cs - k2 * sn, ko2 = k2 * cs + k1 * sn;
    int h1 = j >> 6,        d1 = j & 63;
    int h2 = (j + 512) >> 6, d2 = (j + 512) & 63;
    qbf[((size_t)(b * NH_ + h1) * SQ_ + s) * DH_ + d1] = (bf16)qo1;
    qbf[((size_t)(b * NH_ + h2) * SQ_ + s) * DH_ + d2] = (bf16)qo2;
    size_t ka = ((size_t)(b * NH_ + h1) * SKV_ + PAST_ + s) * DH_ + d1;
    size_t kb = ((size_t)(b * NH_ + h2) * SKV_ + PAST_ + s) * DH_ + d2;
    newK[ka] = ko1;      newK[kb] = ko2;
    newV[ka] = vr[j];    newV[kb] = vr[j + 512];
}

// ---------------------------------------------------- KV cache copy (B128)
__global__ void copy_past(const float4* __restrict__ pk, const float4* __restrict__ pv,
                          float4* __restrict__ nk, float4* __restrict__ nv) {
    int y = blockIdx.y;                 // mat*128 + bh
    int mat = y >> 7, bh = y & 127;
    const float4* src = (mat ? pv : pk) + (size_t)bh * (PAST_ * DH_ / 4);
    float4*       dst = (mat ? nv : nk) + (size_t)bh * (SKV_ * DH_ / 4);
    int t = blockIdx.x * blockDim.x + threadIdx.x;     // 0..65535
    dst[t] = src[t];
}

// ---------------------------------------------------- flash attention partial
__global__ void __launch_bounds__(32)
attn_partial(const bf16* __restrict__ qbf, const float* __restrict__ newK,
             const float* __restrict__ newV, const float* __restrict__ amask,
             float* __restrict__ pctx, float* __restrict__ pm, float* __restrict__ pl) {
    __shared__ bf16 p_lds[16 * 32];     // P tile, row-major 16x32
    __shared__ bf16 vT[64 * 32];        // V chunk transposed: [d][key]
    int bh  = blockIdx.x / NSEG_;
    int seg = blockIdx.x % NSEG_;
    int b   = bh >> 4;
    int lane = threadIdx.x;
    int half = lane >> 4, l16 = lane & 15;

    // Q A-operands (d 0..31 and 32..63), 16x32 bf16 lane layout
    const bf16* qrow = qbf + ((size_t)bh * SQ_ + l16) * DH_;
    union { v16bf v; v8bf h[2]; } aq0, aq1;
    aq0.h[0] = *(const v8bf*)(qrow + half * 16);
    aq0.h[1] = *(const v8bf*)(qrow + half * 16 + 8);
    aq1.h[0] = *(const v8bf*)(qrow + 32 + half * 16);
    aq1.h[1] = *(const v8bf*)(qrow + 40 + half * 16);

    v8f acc0 = {}, acc1 = {}, acc2 = {}, acc3 = {};
    float rowmax[8], rowsum[8];
#pragma unroll
    for (int r = 0; r < 8; ++r) { rowmax[r] = -1e30f; rowsum[r] = 0.f; }

    const float* Kb = newK + (size_t)bh * SKV_ * DH_;
    const float* Vb = newV + (size_t)bh * SKV_ * DH_;
    const int nchunks = (SKV_ + 31) / 32;  // 129

    for (int c = seg; c < nchunks; c += NSEG_) {
        int key0 = c * 32;
        // ---- stage V chunk transposed into LDS (bf16)
        {
            int key = key0 + lane;
            if (key < SKV_) {
                const float* vr = Vb + (size_t)key * DH_;
#pragma unroll
                for (int d = 0; d < DH_; ++d) vT[d * 32 + lane] = (bf16)vr[d];
            } else {
#pragma unroll
                for (int d = 0; d < DH_; ++d) vT[d * 32 + lane] = (bf16)0.f;
            }
        }
        // prefetch next chunk of K for this seg
        {
            int knext = key0 + NSEG_ * 32 + l16;
            if (knext < SKV_) __builtin_prefetch(Kb + (size_t)knext * DH_ + half * 16, 0, 0);
        }
        // ---- scores: two 16x16 tiles, K-dim 64 = 2 WMMA each
        v8f s0 = {}, s1 = {};
        {
            int key = key0 + l16;
            v16bf b0 = {}, b1 = {};
            if (key < SKV_) {
                const float* kp = Kb + (size_t)key * DH_ + half * 16;
#pragma unroll
                for (int e = 0; e < 16; ++e) { b0[e] = (bf16)kp[e]; b1[e] = (bf16)kp[32 + e]; }
            }
            s0 = __builtin_amdgcn_wmma_f32_16x16x32_bf16(false, aq0.v, false, b0, (short)0, s0, false, false);
            s0 = __builtin_amdgcn_wmma_f32_16x16x32_bf16(false, aq1.v, false, b1, (short)0, s0, false, false);
        }
        {
            int key = key0 + 16 + l16;
            v16bf b0 = {}, b1 = {};
            if (key < SKV_) {
                const float* kp = Kb + (size_t)key * DH_ + half * 16;
#pragma unroll
                for (int e = 0; e < 16; ++e) { b0[e] = (bf16)kp[e]; b1[e] = (bf16)kp[32 + e]; }
            }
            s1 = __builtin_amdgcn_wmma_f32_16x16x32_bf16(false, aq0.v, false, b0, (short)0, s1, false, false);
            s1 = __builtin_amdgcn_wmma_f32_16x16x32_bf16(false, aq1.v, false, b1, (short)0, s1, false, false);
        }
        // ---- scale + mask + online softmax
        const float scale = 0.125f;
        int keyA = key0 + l16;
        int keyB = key0 + 16 + l16;
#pragma unroll
        for (int r = 0; r < 8; ++r) {
            int m = r + half * 8;
            float sa = s0[r] * scale, sb = s1[r] * scale;
            if (keyA >= PAST_ && keyA < SKV_) sa += amask[(size_t)b * 256 + m * 16 + (keyA - PAST_)];
            if (keyA >= SKV_) sa = -1e30f;
            if (keyB >= PAST_ && keyB < SKV_) sb += amask[(size_t)b * 256 + m * 16 + (keyB - PAST_)];
            if (keyB >= SKV_) sb = -1e30f;

            float v = fmaxf(sa, sb);
            v = fmaxf(v, __shfl_xor(v, 1, 32));
            v = fmaxf(v, __shfl_xor(v, 2, 32));
            v = fmaxf(v, __shfl_xor(v, 4, 32));
            v = fmaxf(v, __shfl_xor(v, 8, 32));
            float mnew = fmaxf(rowmax[r], v);
            float f  = __expf(rowmax[r] - mnew);
            float p0 = __expf(sa - mnew);
            float p1 = __expf(sb - mnew);
            float ps = p0 + p1;
            ps += __shfl_xor(ps, 1, 32);
            ps += __shfl_xor(ps, 2, 32);
            ps += __shfl_xor(ps, 4, 32);
            ps += __shfl_xor(ps, 8, 32);
            rowsum[r] = rowsum[r] * f + ps;
            rowmax[r] = mnew;
            acc0[r] *= f; acc1[r] *= f; acc2[r] *= f; acc3[r] *= f;
            p_lds[m * 32 + l16]      = (bf16)p0;
            p_lds[m * 32 + 16 + l16] = (bf16)p1;
        }
        __syncthreads();
        // ---- ctx accumulate: P(16x32) x V(32x64) as 4 WMMA
        union { v16bf v; v8bf h[2]; } ap;
        ap.h[0] = *(const v8bf*)(&p_lds[l16 * 32 + half * 16]);
        ap.h[1] = *(const v8bf*)(&p_lds[l16 * 32 + half * 16 + 8]);
        v16bf bv0 = *(const v16bf*)(&vT[(0 * 16 + l16) * 32 + half * 16]);
        v16bf bv1 = *(const v16bf*)(&vT[(1 * 16 + l16) * 32 + half * 16]);
        v16bf bv2 = *(const v16bf*)(&vT[(2 * 16 + l16) * 32 + half * 16]);
        v16bf bv3 = *(const v16bf*)(&vT[(3 * 16 + l16) * 32 + half * 16]);
        acc0 = __builtin_amdgcn_wmma_f32_16x16x32_bf16(false, ap.v, false, bv0, (short)0, acc0, false, false);
        acc1 = __builtin_amdgcn_wmma_f32_16x16x32_bf16(false, ap.v, false, bv1, (short)0, acc1, false, false);
        acc2 = __builtin_amdgcn_wmma_f32_16x16x32_bf16(false, ap.v, false, bv2, (short)0, acc2, false, false);
        acc3 = __builtin_amdgcn_wmma_f32_16x16x32_bf16(false, ap.v, false, bv3, (short)0, acc3, false, false);
        __syncthreads();
    }

    // ---- store partials
    float* ctxp = pctx + ((size_t)(seg * 128 + bh) * 16) * 64;
#pragma unroll
    for (int r = 0; r < 8; ++r) {
        int m = r + half * 8;
        ctxp[(size_t)m * 64 + 0  + l16] = acc0[r];
        ctxp[(size_t)m * 64 + 16 + l16] = acc1[r];
        ctxp[(size_t)m * 64 + 32 + l16] = acc2[r];
        ctxp[(size_t)m * 64 + 48 + l16] = acc3[r];
    }
    if (l16 == 0) {
#pragma unroll
        for (int r = 0; r < 8; ++r) {
            int m = r + half * 8;
            pm[(size_t)(seg * 128 + bh) * 16 + m] = rowmax[r];
            pl[(size_t)(seg * 128 + bh) * 16 + m] = rowsum[r];
        }
    }
}

// ---------------------------------------------------- flash combine
__global__ void combine(const float* __restrict__ pctx, const float* __restrict__ pm,
                        const float* __restrict__ pl, bf16* __restrict__ ctxbf) {
    int bh = blockIdx.x;           // 128
    int d  = threadIdx.x;          // 64
    int b = bh >> 4, h = bh & 15;
    for (int m = 0; m < 16; ++m) {
        float M = -1e30f;
        for (int s = 0; s < NSEG_; ++s)
            M = fmaxf(M, pm[(size_t)(s * 128 + bh) * 16 + m]);
        float L = 0.f, val = 0.f;
        for (int s = 0; s < NSEG_; ++s) {
            float w = __expf(pm[(size_t)(s * 128 + bh) * 16 + m] - M);
            L   += w * pl[(size_t)(s * 128 + bh) * 16 + m];
            val += w * pctx[((size_t)(s * 128 + bh) * 16 + m) * 64 + d];
        }
        int row = b * 16 + m;
        ctxbf[(size_t)row * HID_ + h * 64 + d] = (bf16)(val / L);
    }
}

// ---------------------------------------------------- launch
extern "C" void kernel_launch(void* const* d_in, const int* in_sizes, int n_in,
                              void* d_out, int out_size, void* d_ws, size_t ws_size,
                              hipStream_t stream) {
    (void)in_sizes; (void)n_in; (void)out_size; (void)ws_size;
    const float* hid   = (const float*)d_in[0];
    const float* am    = (const float*)d_in[1];
    const float* pk    = (const float*)d_in[2];
    const float* pv    = (const float*)d_in[3];
    const int*   pos   = (const int*)d_in[4];
    const float* Wq = (const float*)d_in[5];  const float* bq = (const float*)d_in[6];
    const float* Wk = (const float*)d_in[7];  const float* bk = (const float*)d_in[8];
    const float* Wv = (const float*)d_in[9];  const float* bvv = (const float*)d_in[10];
    const float* Wo = (const float*)d_in[11]; const float* bo = (const float*)d_in[12];

    float* outp = (float*)d_out;
    float* newK = outp + (size_t)MROWS_ * HID_;                    // 131072
    float* newV = newK + (size_t)B_ * NH_ * SKV_ * DH_;            // +33685504

    char* w = (char*)d_ws;
    bf16* hb   = (bf16*)w;  w += 262144;
    bf16* WqT  = (bf16*)w;  w += 2097152;
    bf16* WkT  = (bf16*)w;  w += 2097152;
    bf16* WvT  = (bf16*)w;  w += 2097152;
    bf16* WoT  = (bf16*)w;  w += 2097152;
    float* Qraw = (float*)w; w += 524288;
    float* Kraw = (float*)w; w += 524288;
    float* Vraw = (float*)w; w += 524288;
    bf16* qbf  = (bf16*)w;  w += 262144;
    float* pctx = (float*)w; w += 4194304;
    float* pmv  = (float*)w; w += 65536;
    float* plv  = (float*)w; w += 65536;
    bf16* ctxbf = (bf16*)w; w += 262144;

    copy_past<<<dim3(256, 256), dim3(256), 0, stream>>>(
        (const float4*)pk, (const float4*)pv, (float4*)newK, (float4*)newV);
    cvt_hidden<<<512, 256, 0, stream>>>(hid, hb, MROWS_ * HID_);
    cvt_wT<<<dim3(4096, 4), 256, 0, stream>>>(Wq, Wk, Wv, Wo, WqT, WkT, WvT, WoT);
    gemm_bf16<<<512, 32, 0, stream>>>(hb, WqT, bq, Qraw);
    gemm_bf16<<<512, 32, 0, stream>>>(hb, WkT, bk, Kraw);
    gemm_bf16<<<512, 32, 0, stream>>>(hb, WvT, bvv, Vraw);
    rope_scatter<<<256, 256, 0, stream>>>(Qraw, Kraw, Vraw, pos, qbf, newK, newV);
    attn_partial<<<B_ * NH_ * NSEG_, 32, 0, stream>>>(qbf, newK, newV, am, pctx, pmv, plv);
    combine<<<128, 64, 0, stream>>>(pctx, pmv, plv, ctxbf);
    gemm_bf16<<<512, 32, 0, stream>>>(ctxbf, WoT, bo, outp);
}
